// PEAttention_2164663517453
// MI455X (gfx1250) — matrix operand order repaired and checked by reading, not compiled, not run on hardware
//
#include <hip/hip_runtime.h>
#include <hip/hip_bf16.h>

typedef __bf16 bf16;
typedef __bf16 v16bf __attribute__((ext_vector_type(16)));
typedef float  v8f   __attribute__((ext_vector_type(8)));
typedef float  v4f   __attribute__((ext_vector_type(4)));

#define WMMA_BF16(a, b, c) \
  __builtin_amdgcn_wmma_f32_16x16x32_bf16(false, (a), false, (b), (short)0, (c), false, false)

// Problem dims
#define BB 2
#define LL 2048
#define DM 1024
#define HH 16
#define DH 64
#define MR 4096            // B*L
#define PB_ROW 2056        // LDS P row length in bf16 elems (2048 + 8 pad -> 16B pad, bank-conflict free)
#define SMEM_BYTES (8192 + 4 * 16 * PB_ROW * 2)   // pb table + 4 waves * 16 rows

// ---- WMMA operand loaders (CDNA5 §7.12.2 layouts) -------------------------
// A (16x32 bf16): lane&15 = M row; low half K 0..7 & 16..23, high half K 8..15 & 24..31
__device__ __forceinline__ v16bf load_a16(const bf16* rowp, int lane) {
  const char* p = (const char*)rowp + ((lane & 16) ? 16 : 0);
  v16bf r;
  ((v4f*)&r)[0] = *(const v4f*)(p);
  ((v4f*)&r)[1] = *(const v4f*)(p + 32);
  return r;
}
// B (32x16 bf16): lane&15 = N col; low half K 0..7 & 16..23, high half K 8..15 & 24..31
__device__ __forceinline__ v16bf load_b16(const bf16* rowp, int lane) {
  const char* p = (const char*)rowp + ((lane & 16) ? 16 : 0);
  v16bf r;
  ((v4f*)&r)[0] = *(const v4f*)(p);
  ((v4f*)&r)[1] = *(const v4f*)(p + 32);
  return r;
}

// ---- Kernel 1: convert x -> bf16, W{q,k,v,o} -> bf16 transposed ------------
__global__ void prep_kernel(const float* __restrict__ x,
                            const float* __restrict__ Wq, const float* __restrict__ Wk,
                            const float* __restrict__ Wv, const float* __restrict__ Wo,
                            bf16* __restrict__ xb, bf16* __restrict__ Wt4) {
  size_t tid    = (size_t)blockIdx.x * blockDim.x + threadIdx.x;
  size_t stride = (size_t)gridDim.x * blockDim.x;
  for (size_t i = tid; i < (size_t)MR * DM; i += stride) xb[i] = (bf16)x[i];
  const float* Ws[4] = {Wq, Wk, Wv, Wo};
  for (int w = 0; w < 4; ++w) {
    const float* W  = Ws[w];
    bf16*        Wt = Wt4 + (size_t)w * DM * DM;
    for (size_t i = tid; i < (size_t)DM * DM; i += stride) {
      size_t n = i >> 10, kk = i & 1023;          // Wt[n,k] = W[k,n]
      Wt[i] = (bf16)W[kk * DM + n];
    }
  }
}

// ---- Kernel 2: fused QKV projection + bias + RoPE --------------------------
// grid(256 row-tiles, 16 heads, 3={q,k,v}), 1 wave per block, 16x64 tile/wave
__global__ void __launch_bounds__(32) qkv_kernel(
    const bf16* __restrict__ xb, const bf16* __restrict__ Wt3,
    const float* __restrict__ bq, const float* __restrict__ bk, const float* __restrict__ bv,
    bf16* __restrict__ qO, bf16* __restrict__ kO, bf16* __restrict__ vtO) {
  const int lane = threadIdx.x & 31;
  const int row  = lane & 15;
  const int hi   = (lane >> 4) & 1;
  const int mBase = blockIdx.x * 16;
  const int h = blockIdx.y;
  const int z = blockIdx.z;                       // 0=q 1=k 2=v

  const bf16*  Wt   = Wt3 + (size_t)z * DM * DM;
  const float* bias = (z == 0) ? bq : (z == 1) ? bk : bv;

  v8f zero8 = {0.f,0.f,0.f,0.f,0.f,0.f,0.f,0.f};
  v8f acc[4];
  for (int nt = 0; nt < 4; ++nt) acc[nt] = zero8;

  const bf16* arow = xb + (size_t)(mBase + row) * DM;
  for (int kc = 0; kc < DM; kc += 32) {
    v16bf a = load_a16(arow + kc, lane);
#pragma unroll
    for (int nt = 0; nt < 4; ++nt) {
      int n = h * DH + nt * 16 + row;
      v16bf b = load_b16(Wt + (size_t)n * DM + kc, lane);
      acc[nt] = WMMA_BF16(a, b, acc[nt]);
    }
  }

#pragma unroll
  for (int nt = 0; nt < 4; ++nt) {
    int   d    = nt * 16 + row;                   // head-dim index 0..63
    float bv_  = bias[h * DH + d];
    float freq = __powf(10000.0f, -(float)(2 * (d >> 1)) / 64.0f);
#pragma unroll
    for (int r = 0; r < 8; ++r) {
      int   m   = mBase + r + hi * 8;             // row in [0, B*L)
      float val = acc[nt][r] + bv_;
      int   pos = m & (LL - 1);
      if (z < 2) {                                // RoPE on q,k (pairs = adjacent lanes)
        float other = __shfl_xor(val, 1, 32);
        float th = (float)pos * freq, s, c;
        __sincosf(th, &s, &c);
        val = (d & 1) ? (other * s + val * c) : (val * c - other * s);
      }
      int    b_  = m >> 11;
      size_t bh  = (size_t)(b_ * HH + h);
      if (z == 2) vtO[(bh * DH + d) * LL + pos] = (bf16)val;   // V transposed [B,H,Dh,L]
      else {
        bf16* o = (z == 0) ? qO : kO;
        o[(bh * LL + pos) * DH + d] = (bf16)val;               // [B,H,L,Dh]
      }
    }
  }
}

// ---- Kernel 3: fused attention -------------------------------------------
// grid(B*H=32, L/64=32), 4 waves/block; wave owns a 16-row query tile.
extern __shared__ __align__(16) char smem[];
__global__ void __launch_bounds__(128) attn_kernel(
    const bf16* __restrict__ q, const bf16* __restrict__ k, const bf16* __restrict__ vt,
    float* __restrict__ p_out, bf16* __restrict__ o_buf) {
  float* pb = (float*)smem;                       // periodic bias table pb[|i-j|]
  const int tid = threadIdx.x, lane = tid & 31, w = tid >> 5;
  for (int d0 = tid; d0 < LL; d0 += 128) {
    float d = (float)d0;
    pb[d0] = 0.5f  * __cosf(6.2831853f * d / 24.0f)
           + 0.25f * __cosf(6.2831853f * d / 168.0f);
  }
  __syncthreads();
  bf16* P = (bf16*)(smem + 8192) + (size_t)w * 16 * PB_ROW;

  const int bh    = blockIdx.x;                   // b*H + h
  const int itile = blockIdx.y * 4 + w;
  const int iBase = itile * 16;
  const int row   = lane & 15;
  const int hi    = (lane >> 4) & 1;
  const float scale = 0.125f;                     // 1/sqrt(64)
  v8f zero8 = {0.f,0.f,0.f,0.f,0.f,0.f,0.f,0.f};

  const bf16* qrow = q + ((size_t)bh * LL + iBase + row) * DH;
  v16bf aq0 = load_a16(qrow, lane);
  v16bf aq1 = load_a16(qrow + 32, lane);

  float mx[8], ssum[8];
#pragma unroll
  for (int r = 0; r < 8; ++r) { mx[r] = -3.0e38f; ssum[r] = 0.0f; }

  // ---- pass 1: online (max,sum) of softmax logits --------------------------
  for (int jt = 0; jt <= itile; ++jt) {
    const int   jBase = jt * 16;
    const bf16* krow  = k + ((size_t)bh * LL + jBase + row) * DH;
    v16bf b0 = load_b16(krow, lane);
    v16bf b1 = load_b16(krow + 32, lane);
    v8f c = zero8;
    c = WMMA_BF16(aq0, b0, c);
    c = WMMA_BF16(aq1, b1, c);
    int j = jBase + row;
#pragma unroll
    for (int r = 0; r < 8; ++r) {
      int   i   = iBase + r + hi * 8;
      float val = (j <= i) ? (c[r] * scale + pb[i - j]) : -3.0e38f;  // alibi==0 for j<=i
      float mn  = fmaxf(mx[r], val);
      ssum[r] = ssum[r] * __expf(mx[r] - mn) + __expf(val - mn);
      mx[r] = mn;
    }
  }
  // reduce stats across the 16 lanes of each half (rows r / r+8)
  for (int off = 1; off < 16; off <<= 1) {
#pragma unroll
    for (int r = 0; r < 8; ++r) {
      float m2 = __shfl_xor(mx[r], off, 32);
      float s2 = __shfl_xor(ssum[r], off, 32);
      float mn = fmaxf(mx[r], m2);
      ssum[r] = ssum[r] * __expf(mx[r] - mn) + s2 * __expf(m2 - mn);
      mx[r] = mn;
    }
  }
  float invs[8];
#pragma unroll
  for (int r = 0; r < 8; ++r) invs[r] = 1.0f / ssum[r];

  // ---- pass 2: recompute, write normalized p once, stash exp in LDS --------
  float* prow  = p_out + ((size_t)bh * LL + iBase) * LL;
  const int jtEnd = itile | 1;                    // round tiles up to 32-col multiple
  for (int jt = 0; jt <= jtEnd; ++jt) {
    const int   jBase = jt * 16;
    const bf16* krow  = k + ((size_t)bh * LL + jBase + row) * DH;
    v16bf b0 = load_b16(krow, lane);
    v16bf b1 = load_b16(krow + 32, lane);
    v8f c = zero8;
    c = WMMA_BF16(aq0, b0, c);
    c = WMMA_BF16(aq1, b1, c);
    int j = jBase + row;
#pragma unroll
    for (int r = 0; r < 8; ++r) {
      int   i   = iBase + r + hi * 8;
      float val = (j <= i) ? (c[r] * scale + pb[i - j]) : -3.0e38f;
      float pu  = __expf(val - mx[r]);
      int   lr  = r + hi * 8;
      P[(size_t)lr * PB_ROW + j]    = (bf16)pu;     // unnormalized, bf16, LDS
      prow[(size_t)lr * LL + j]     = pu * invs[r]; // normalized p to HBM (written once)
    }
  }
  // zero the remaining upper-triangle tiles of p
  for (int jt = jtEnd + 1; jt < LL / 16; ++jt) {
    int j = jt * 16 + row;
#pragma unroll
    for (int r = 0; r < 8; ++r) prow[(size_t)(r + hi * 8) * LL + j] = 0.0f;
  }

  // ---- pass 3: out = (P_unnorm @ V) * invs ---------------------------------
  v8f o[4];
  for (int nt = 0; nt < 4; ++nt) o[nt] = zero8;
  const int   nch  = (jtEnd + 1) >> 1;            // K chunks of 32 over j
  const bf16* Prow = P + (size_t)row * PB_ROW;    // A from LDS
  for (int c32 = 0; c32 < nch; ++c32) {
    v16bf a = load_a16(Prow + c32 * 32, lane);
#pragma unroll
    for (int nt = 0; nt < 4; ++nt) {
      int d = nt * 16 + row;
      const bf16* vrow = vt + ((size_t)bh * DH + d) * LL + c32 * 32;
      v16bf b = load_b16(vrow, lane);
      o[nt] = WMMA_BF16(a, b, o[nt]);
    }
  }
  const int b_ = bh >> 4, h_ = bh & 15;
#pragma unroll
  for (int nt = 0; nt < 4; ++nt) {
    int d = nt * 16 + row;
#pragma unroll
    for (int r = 0; r < 8; ++r) {
      int pos = iBase + r + hi * 8;
      o_buf[((size_t)(b_ * LL + pos)) * DM + h_ * DH + d] = (bf16)(o[nt][r] * invs[r]);
    }
  }
}

// ---- Kernel 4: out = o_buf @ Wo + bo (fp32 result) -------------------------
__global__ void __launch_bounds__(32) oproj_kernel(
    const bf16* __restrict__ ob, const bf16* __restrict__ Wot,
    const float* __restrict__ bo, float* __restrict__ out) {
  const int lane = threadIdx.x & 31;
  const int row  = lane & 15;
  const int hi   = (lane >> 4) & 1;
  const int mBase = blockIdx.x * 16;
  const int nBase = blockIdx.y * 64;

  v8f zero8 = {0.f,0.f,0.f,0.f,0.f,0.f,0.f,0.f};
  v8f acc[4];
  for (int nt = 0; nt < 4; ++nt) acc[nt] = zero8;

  const bf16* arow = ob + (size_t)(mBase + row) * DM;
  for (int kc = 0; kc < DM; kc += 32) {
    v16bf a = load_a16(arow + kc, lane);
#pragma unroll
    for (int nt = 0; nt < 4; ++nt) {
      v16bf b = load_b16(Wot + (size_t)(nBase + nt * 16 + row) * DM + kc, lane);
      acc[nt] = WMMA_BF16(a, b, acc[nt]);
    }
  }
#pragma unroll
  for (int nt = 0; nt < 4; ++nt) {
    int   n  = nBase + nt * 16 + row;
    float bb = bo[n];
#pragma unroll
    for (int r = 0; r < 8; ++r)
      out[(size_t)(mBase + r + hi * 8) * DM + n] = acc[nt][r] + bb;
  }
}

// ---- Host launcher ---------------------------------------------------------
extern "C" void kernel_launch(void* const* d_in, const int* in_sizes, int n_in,
                              void* d_out, int out_size, void* d_ws, size_t ws_size,
                              hipStream_t stream) {
  const float* x  = (const float*)d_in[0];
  const float* Wq = (const float*)d_in[1];
  const float* bq = (const float*)d_in[2];
  const float* Wk = (const float*)d_in[3];
  const float* bk = (const float*)d_in[4];
  const float* Wv = (const float*)d_in[5];
  const float* bv = (const float*)d_in[6];
  const float* Wo = (const float*)d_in[7];
  const float* bo = (const float*)d_in[8];

  float* out   = (float*)d_out;                       // [B,L,D] = 4194304 fp32
  float* p_out = out + (size_t)MR * DM;               // [B,H,L,L] fp32

  const size_t E = (size_t)MR * DM;                   // 4194304 elems, used as stride
  bf16* xb  = (bf16*)d_ws;
  bf16* Wt4 = xb  + E;                                // 4 transposed bf16 weights
  bf16* qb  = Wt4 + 4 * (size_t)DM * DM;
  bf16* kb  = qb  + E;
  bf16* vtb = kb  + E;
  bf16* ob  = vtb + E;

  (void)hipFuncSetAttribute((const void*)attn_kernel,
                            hipFuncAttributeMaxDynamicSharedMemorySize, SMEM_BYTES);

  prep_kernel<<<1024, 256, 0, stream>>>(x, Wq, Wk, Wv, Wo, xb, Wt4);
  qkv_kernel<<<dim3(MR / 16, HH, 3), 32, 0, stream>>>(xb, Wt4, bq, bk, bv, qb, kb, vtb);
  attn_kernel<<<dim3(BB * HH, LL / 64), 128, SMEM_BYTES, stream>>>(qb, kb, vtb, p_out, ob);
  oproj_kernel<<<dim3(MR / 16, DM / 64), 32, 0, stream>>>(ob, Wt4 + 3 * (size_t)DM * DM, bo, out);
}
